// QuantizedLinear_1864015806714
// MI455X (gfx1250) — 8-bit matrix operand order repaired and checked by reading, compiled, not run on hardware
//
#include <hip/hip_runtime.h>
#include <stdint.h>

typedef __attribute__((ext_vector_type(8))) int v8i;
typedef int v4i __attribute__((vector_size(16)));

#define GLOBAL_AS __attribute__((address_space(1)))
#define LDS_AS    __attribute__((address_space(3)))

#if defined(__has_builtin)
#  if __has_builtin(__builtin_amdgcn_global_load_async_to_lds_b128) && \
      __has_builtin(__builtin_amdgcn_s_wait_asynccnt)
#    define USE_ASYNC_LDS 1
#  endif
#endif
#ifndef USE_ASYNC_LDS
#  define USE_ASYNC_LDS 0
#endif

#define N_TOK 8192
#define IN_F  4096
#define OUT_F 4096
#define BM 128
#define BN 128
#define BK 64
#define KSTEPS (IN_F / BK)

__global__ __launch_bounds__(256)
void qlinear_i8_wmma(const int8_t* __restrict__ x,
                     const int8_t* __restrict__ w,
                     const int32_t* __restrict__ bias,
                     const float* __restrict__ wscale,
                     float* __restrict__ out)
{
    __shared__ int8_t As[2][BM * BK];
    __shared__ int8_t Bs[2][BN * BK];

    const int t    = threadIdx.x;
    const int lane = t & 31;
    const int wv   = t >> 5;
    const int wr   = wv >> 1;
    const int wc   = wv & 1;
    const int r    = lane & 15;
    const int h    = lane >> 4;

    const int rowBase = blockIdx.y * BM;
    const int colBase = blockIdx.x * BN;

    const int l0    = t * 16;
    const int ldRow = l0 >> 6;
    const int ldCol = l0 & 63;
    const int8_t* gA0 = x + (size_t)(rowBase + ldRow) * IN_F + ldCol;
    const int8_t* gA1 = gA0 + (size_t)64 * IN_F;
    const int8_t* gB0 = w + (size_t)(colBase + ldRow) * IN_F + ldCol;
    const int8_t* gB1 = gB0 + (size_t)64 * IN_F;

#if USE_ASYNC_LDS
    auto cp16 = [](const int8_t* g, const int8_t* l) {
        __builtin_amdgcn_global_load_async_to_lds_b128(
            (GLOBAL_AS v4i*)g, (LDS_AS v4i*)l,  0,  0);
    };
#endif

    v8i acc[2][4];
#pragma unroll
    for (int mt = 0; mt < 2; ++mt)
#pragma unroll
        for (int nt = 0; nt < 4; ++nt)
            acc[mt][nt] = (v8i)(0);

    const int aOff = (wr * 32 + r) * BK;
    const int bOff = (wc * 64 + r) * BK;

    auto compute = [&](const int8_t* as, const int8_t* bs) {
        v8i afrag[2];
#pragma unroll
        for (int mt = 0; mt < 2; ++mt) {
            const int8_t* ap = as + aOff + mt * 16 * BK;
            union { int64_t d[4]; v8i v; } ua;
#pragma unroll
            for (int j = 0; j < 4; ++j)
                ua.d[j] = *(const int64_t*)(ap + 8 * j + 16 * h + 16 * (j >> 1));
            afrag[mt] = ua.v;
        }
        v8i bfrag[4];
#pragma unroll
        for (int nt = 0; nt < 4; ++nt) {
            const int8_t* bp = bs + bOff + nt * 16 * BK;
            union { int4 qd[2]; v8i v; } ub;
            ub.qd[0] = *(const int4*)(bp + 16 * h);
            ub.qd[1] = *(const int4*)(bp + 32 + 16 * h);
            bfrag[nt] = ub.v;
        }
#pragma unroll
        for (int mt = 0; mt < 2; ++mt)
#pragma unroll
            for (int nt = 0; nt < 4; ++nt)
                acc[mt][nt] = __builtin_amdgcn_wmma_i32_16x16x64_iu8(
                     true, afrag[mt],
                     true, bfrag[nt],
                    acc[mt][nt],
                     false,  false);
    };

    const int8_t* curA = &As[0][0];
    const int8_t* nxtA = &As[1][0];
    const int8_t* curB = &Bs[0][0];
    const int8_t* nxtB = &Bs[1][0];

#if USE_ASYNC_LDS
    cp16(gA0, curA + l0);
    cp16(gA1, curA + l0 + 4096);
    cp16(gB0, curB + l0);
    cp16(gB1, curB + l0 + 4096);
    __builtin_amdgcn_s_wait_asynccnt(0);
#else
    *(int4*)(curA + l0)        = *(const int4*)gA0;
    *(int4*)(curA + l0 + 4096) = *(const int4*)gA1;
    *(int4*)(curB + l0)        = *(const int4*)gB0;
    *(int4*)(curB + l0 + 4096) = *(const int4*)gB1;
#endif
    __syncthreads();

#pragma unroll 1
    for (int ks = 0; ks < KSTEPS - 1; ++ks) {
        const int nk = (ks + 1) * BK;
#if USE_ASYNC_LDS
        cp16(gA0 + nk, nxtA + l0);
        cp16(gA1 + nk, nxtA + l0 + 4096);
        cp16(gB0 + nk, nxtB + l0);
        cp16(gB1 + nk, nxtB + l0 + 4096);
        compute(curA, curB);
        __builtin_amdgcn_s_wait_asynccnt(0);
#else
        const int4 ra0 = *(const int4*)(gA0 + nk);
        const int4 ra1 = *(const int4*)(gA1 + nk);
        const int4 rb0 = *(const int4*)(gB0 + nk);
        const int4 rb1 = *(const int4*)(gB1 + nk);
        compute(curA, curB);
        *(int4*)(nxtA + l0)        = ra0;
        *(int4*)(nxtA + l0 + 4096) = ra1;
        *(int4*)(nxtB + l0)        = rb0;
        *(int4*)(nxtB + l0 + 4096) = rb1;
#endif
        __syncthreads();
        const int8_t* tA = curA; curA = nxtA; nxtA = tA;
        const int8_t* tB = curB; curB = nxtB; nxtB = tB;
    }

    compute(curA, curB);

#pragma unroll
    for (int nt = 0; nt < 4; ++nt) {
        const int col = colBase + wc * 64 + nt * 16 + r;
        const float sc = 0.4f * wscale[col];
        const int   bz = bias[col];
#pragma unroll
        for (int mt = 0; mt < 2; ++mt) {
            const int row0 = rowBase + wr * 32 + mt * 16 + 8 * h;
            v8i a = acc[mt][nt];
#pragma unroll
            for (int g = 0; g < 8; ++g) {
                float f = (float)(a[g] + bz) * sc;
                f = rintf(f);
                f = fminf(fmaxf(f, -128.0f), 127.0f);
                out[(size_t)(row0 + g) * OUT_F + col] = f;
            }
        }
    }
}

extern "C" void kernel_launch(void* const* d_in, const int* in_sizes, int n_in,
                              void* d_out, int out_size, void* d_ws, size_t ws_size,
                              hipStream_t stream) {
    const int8_t*  x  = (const int8_t*)d_in[0];
    const int8_t*  w  = (const int8_t*)d_in[1];
    const int32_t* b  = (const int32_t*)d_in[2];
    const float*   ws = (const float*)d_in[3];
    float* out = (float*)d_out;

    dim3 grid(OUT_F / BN, N_TOK / BM);
    qlinear_i8_wmma<<<grid, dim3(256), 0, stream>>>(x, w, b, ws, out);
}
